// MultiheadAttention_8117488190286
// MI455X (gfx1250) — matrix operand order repaired and checked by reading, not compiled, not run on hardware
//
#include <hip/hip_runtime.h>

typedef __attribute__((ext_vector_type(16))) __bf16 v16bf;
typedef __attribute__((ext_vector_type(8)))  __bf16 v8bf;
typedef __attribute__((ext_vector_type(8)))  float  v8f;

#define BATCH  4
#define HEADS  16
#define SEQ    2048
#define DHEAD  64
#define DMODEL 1024

static __device__ __forceinline__ v8f wmma_bf16(v16bf a, v16bf b, v8f c) {
  // D = A(16x32 bf16) * B(32x16 bf16) + C(16x16 f32)
  return __builtin_amdgcn_wmma_f32_16x16x32_bf16(false, a, false, b, (short)0, c, false, false);
}

// ---------------------------------------------------------------------------
// Kernel 0: convert W_o (fp32 [1024][1024], row = out-feature n, col = k) to bf16
// ---------------------------------------------------------------------------
__global__ __launch_bounds__(256) void wcvt_kernel(const float* __restrict__ W,
                                                   __bf16* __restrict__ Wb) {
  int i = (blockIdx.x * 256 + threadIdx.x) * 4;
  float4 f = *(const float4*)(W + i);
  Wb[i + 0] = (__bf16)f.x;
  Wb[i + 1] = (__bf16)f.y;
  Wb[i + 2] = (__bf16)f.z;
  Wb[i + 3] = (__bf16)f.w;
}

// ---------------------------------------------------------------------------
// Kernel 1: flash attention.  grid = (SEQ/128, BATCH*HEADS), block = 256 (8 waves)
// 64-key tiles; per tile: 8 WMMA (QK^T) + 2 WMMA (row sums) + 8 WMMA (PV).
// Row sums computed as P @ ones via WMMA (consistent with bf16-rounded P).
// X output layout: [b][s][h*64 + d]  (== projection GEMM input, bf16)
// ---------------------------------------------------------------------------
__global__ __launch_bounds__(256) void flash_kernel(const float* __restrict__ Q,
                                                    const float* __restrict__ K,
                                                    const float* __restrict__ V,
                                                    __bf16* __restrict__ X) {
  const int bh   = blockIdx.y;
  const int b    = bh >> 4;
  const int h    = bh & 15;
  const int q0   = blockIdx.x * 128;
  const int tid  = threadIdx.x;
  const int w    = tid >> 5;
  const int lane = tid & 31;
  const int half = (lane >> 4) & 1;
  const int l16  = lane & 15;

  __shared__ __align__(32) __bf16 Kt[64][64];      // [key][dh]       8 KB
  __shared__ __align__(32) __bf16 Vt[64][64];      // [dh][key]       8 KB (transposed)
  __shared__ __align__(32) __bf16 Pb[8][16][64];   // per-wave P     16 KB

  const size_t head_base = (size_t)bh * SEQ * DHEAD;

  // --- resident Q A-fragments (dh 0..31 and 32..63) ---
  v16bf a0, a1;
  {
    const float* qrow = Q + head_base + (size_t)(q0 + w * 16 + l16) * DHEAD;
    const int koff = half * 16;
#pragma unroll
    for (int j = 0; j < 8; ++j) {
      a0[j]     = (__bf16)qrow[koff + j];
      a0[8 + j] = (__bf16)qrow[koff + 8 + j];
      a1[j]     = (__bf16)qrow[32 + koff + j];
      a1[8 + j] = (__bf16)qrow[32 + koff + 8 + j];
    }
  }

  // all-ones B fragment (32x16) for row-sum WMMA
  v16bf vones;
#pragma unroll
  for (int j = 0; j < 16; ++j) vones[j] = (__bf16)1.0f;

  v8f o0 = {}, o1 = {}, o2 = {}, o3 = {};
  float mrun[8], lrun[8], alph[8];
#pragma unroll
  for (int j = 0; j < 8; ++j) { mrun[j] = -3.402823466e+38f; lrun[j] = 0.0f; }

  // cooperative-staging slot: one key row, 16 dh values (4 float4 loads)
  const int    skey = tid >> 2;        // 0..63
  const int    sd0  = (tid & 3) * 16;  // 0,16,32,48
  const float* kgp  = K + head_base + (size_t)skey * DHEAD + sd0;
  const float* vgp  = V + head_base + (size_t)skey * DHEAD + sd0;

  for (int kt0 = 0; kt0 < SEQ; kt0 += 64) {
    // ---- stage K tile (row-major) and V tile (transposed) as bf16 in LDS ----
    {
      const float* kp = kgp + (size_t)kt0 * DHEAD;
      const float* vp = vgp + (size_t)kt0 * DHEAD;
      float4 kf[4], vf[4];
#pragma unroll
      for (int i = 0; i < 4; ++i) { kf[i] = ((const float4*)kp)[i]; vf[i] = ((const float4*)vp)[i]; }
      v8bf klo, khi;
      klo[0] = (__bf16)kf[0].x; klo[1] = (__bf16)kf[0].y; klo[2] = (__bf16)kf[0].z; klo[3] = (__bf16)kf[0].w;
      klo[4] = (__bf16)kf[1].x; klo[5] = (__bf16)kf[1].y; klo[6] = (__bf16)kf[1].z; klo[7] = (__bf16)kf[1].w;
      khi[0] = (__bf16)kf[2].x; khi[1] = (__bf16)kf[2].y; khi[2] = (__bf16)kf[2].z; khi[3] = (__bf16)kf[2].w;
      khi[4] = (__bf16)kf[3].x; khi[5] = (__bf16)kf[3].y; khi[6] = (__bf16)kf[3].z; khi[7] = (__bf16)kf[3].w;
      *(v8bf*)&Kt[skey][sd0]     = klo;
      *(v8bf*)&Kt[skey][sd0 + 8] = khi;
      float vv[16] = {vf[0].x, vf[0].y, vf[0].z, vf[0].w, vf[1].x, vf[1].y, vf[1].z, vf[1].w,
                      vf[2].x, vf[2].y, vf[2].z, vf[2].w, vf[3].x, vf[3].y, vf[3].z, vf[3].w};
#pragma unroll
      for (int i = 0; i < 16; ++i) Vt[sd0 + i][skey] = (__bf16)vv[i];
      if (kt0 + 64 < SEQ) {                 // pull next tile toward L2/L0
        __builtin_prefetch(kp + 64 * DHEAD, 0, 1);
        __builtin_prefetch(vp + 64 * DHEAD, 0, 1);
      }
    }
    __syncthreads();

    // ---- scores: S[16 x 64] = Q * K^T, four 16x16 WMMA tiles, K-dim = 64 ----
    v8f s0 = {}, s1 = {}, s2 = {}, s3 = {};
    {
      v16bf bk;
      bk = *(const v16bf*)&Kt[l16][half * 16];            s0 = wmma_bf16(a0, bk, s0);
      bk = *(const v16bf*)&Kt[l16][32 + half * 16];       s0 = wmma_bf16(a1, bk, s0);
      bk = *(const v16bf*)&Kt[16 + l16][half * 16];       s1 = wmma_bf16(a0, bk, s1);
      bk = *(const v16bf*)&Kt[16 + l16][32 + half * 16];  s1 = wmma_bf16(a1, bk, s1);
      bk = *(const v16bf*)&Kt[32 + l16][half * 16];       s2 = wmma_bf16(a0, bk, s2);
      bk = *(const v16bf*)&Kt[32 + l16][32 + half * 16];  s2 = wmma_bf16(a1, bk, s2);
      bk = *(const v16bf*)&Kt[48 + l16][half * 16];       s3 = wmma_bf16(a0, bk, s3);
      bk = *(const v16bf*)&Kt[48 + l16][32 + half * 16];  s3 = wmma_bf16(a1, bk, s3);
    }

    // ---- online softmax (fp32).  Acc slot j -> row j + 8*half; col = l16. ----
    const float scale = 0.125f;  // 1/sqrt(64)
#pragma unroll
    for (int j = 0; j < 8; ++j) {
      float x0 = s0[j] * scale;
      float x1 = s1[j] * scale;
      float x2 = s2[j] * scale;
      float x3 = s3[j] * scale;
      float tm = fmaxf(fmaxf(x0, x1), fmaxf(x2, x3));
      tm = fmaxf(tm, __shfl_xor(tm, 1));
      tm = fmaxf(tm, __shfl_xor(tm, 2));
      tm = fmaxf(tm, __shfl_xor(tm, 4));
      tm = fmaxf(tm, __shfl_xor(tm, 8));
      float mn = fmaxf(mrun[j], tm);
      alph[j]  = __expf(mrun[j] - mn);
      mrun[j]  = mn;
      float p0 = __expf(x0 - mn);
      float p1 = __expf(x1 - mn);
      float p2 = __expf(x2 - mn);
      float p3 = __expf(x3 - mn);
      o0[j] *= alph[j]; o1[j] *= alph[j]; o2[j] *= alph[j]; o3[j] *= alph[j];
      // C-layout -> LDS (row-major P) for A-layout reload
      Pb[w][half * 8 + j][l16]      = (__bf16)p0;
      Pb[w][half * 8 + j][16 + l16] = (__bf16)p1;
      Pb[w][half * 8 + j][32 + l16] = (__bf16)p2;
      Pb[w][half * 8 + j][48 + l16] = (__bf16)p3;
    }
    __builtin_amdgcn_wave_barrier();   // order per-wave LDS store->load

    // ---- reload P as two A-fragments (16 x 32 bf16 each) ----
    v16bf pa0, pa1;
    {
      v8bf lo0 = *(const v8bf*)&Pb[w][l16][half * 16];
      v8bf hi0 = *(const v8bf*)&Pb[w][l16][half * 16 + 8];
      v8bf lo1 = *(const v8bf*)&Pb[w][l16][32 + half * 16];
      v8bf hi1 = *(const v8bf*)&Pb[w][l16][32 + half * 16 + 8];
#pragma unroll
      for (int j = 0; j < 8; ++j) {
        pa0[j] = lo0[j]; pa0[8 + j] = hi0[j];
        pa1[j] = lo1[j]; pa1[8 + j] = hi1[j];
      }
    }

    // ---- row sums via WMMA: every column of (P @ ones) is the row sum ----
    {
      v8f ssum = {};
      ssum = wmma_bf16(pa0, vones, ssum);
      ssum = wmma_bf16(pa1, vones, ssum);
#pragma unroll
      for (int j = 0; j < 8; ++j) lrun[j] = lrun[j] * alph[j] + ssum[j];
    }

    // ---- O += P * V : 4 dh tiles x 2 key halves ----
    {
      v16bf bv;
      bv = *(const v16bf*)&Vt[l16][half * 16];            o0 = wmma_bf16(pa0, bv, o0);
      bv = *(const v16bf*)&Vt[l16][32 + half * 16];       o0 = wmma_bf16(pa1, bv, o0);
      bv = *(const v16bf*)&Vt[16 + l16][half * 16];       o1 = wmma_bf16(pa0, bv, o1);
      bv = *(const v16bf*)&Vt[16 + l16][32 + half * 16];  o1 = wmma_bf16(pa1, bv, o1);
      bv = *(const v16bf*)&Vt[32 + l16][half * 16];       o2 = wmma_bf16(pa0, bv, o2);
      bv = *(const v16bf*)&Vt[32 + l16][32 + half * 16];  o2 = wmma_bf16(pa1, bv, o2);
      bv = *(const v16bf*)&Vt[48 + l16][half * 16];       o3 = wmma_bf16(pa0, bv, o3);
      bv = *(const v16bf*)&Vt[48 + l16][32 + half * 16];  o3 = wmma_bf16(pa1, bv, o3);
    }
    __syncthreads();   // protect Kt/Vt before next stage
  }

  // ---- epilogue: O /= l, write bf16 X[b][s][h*64 + d] ----
#pragma unroll
  for (int j = 0; j < 8; ++j) {
    const float inv  = 1.0f / lrun[j];
    const int   qrow = q0 + w * 16 + half * 8 + j;
    __bf16* xp = X + (size_t)(b * SEQ + qrow) * DMODEL + h * DHEAD + l16;
    xp[0]  = (__bf16)(o0[j] * inv);
    xp[16] = (__bf16)(o1[j] * inv);
    xp[32] = (__bf16)(o2[j] * inv);
    xp[48] = (__bf16)(o3[j] * inv);
  }
}

// ---------------------------------------------------------------------------
// Kernel 2: out[8192,1024] = X[8192,1024] @ W_o^T + b_o   (bf16 WMMA, f32 out)
// grid = (8192/128, 1024/64), block = 256 (8 waves); wave -> 16x64 tile.
// ---------------------------------------------------------------------------
__global__ __launch_bounds__(256) void proj_kernel(const __bf16* __restrict__ X,
                                                   const __bf16* __restrict__ Wb,
                                                   const float* __restrict__ bias,
                                                   float* __restrict__ out) {
  const int tid  = threadIdx.x;
  const int w    = tid >> 5;
  const int lane = tid & 31;
  const int half = (lane >> 4) & 1;
  const int l16  = lane & 15;
  const int m0   = blockIdx.x * 128 + w * 16;
  const int n0   = blockIdx.y * 64;

  v8f acc0 = {}, acc1 = {}, acc2 = {}, acc3 = {};
  const __bf16* xrow = X  + (size_t)(m0 + l16) * DMODEL;
  const __bf16* wrow = Wb + (size_t)(n0 + l16) * DMODEL;

#pragma unroll 2
  for (int k0 = 0; k0 < DMODEL; k0 += 32) {
    v16bf a;
    {
      v8bf lo = *(const v8bf*)(xrow + k0 + half * 16);
      v8bf hi = *(const v8bf*)(xrow + k0 + half * 16 + 8);
#pragma unroll
      for (int j = 0; j < 8; ++j) { a[j] = lo[j]; a[8 + j] = hi[j]; }
    }
    const int kb = k0 + half * 16;
    v16bf b0 = *(const v16bf*)(wrow + kb);
    v16bf b1 = *(const v16bf*)(wrow + 16 * DMODEL + kb);
    v16bf b2 = *(const v16bf*)(wrow + 32 * DMODEL + kb);
    v16bf b3 = *(const v16bf*)(wrow + 48 * DMODEL + kb);
    acc0 = wmma_bf16(a, b0, acc0);
    acc1 = wmma_bf16(a, b1, acc1);
    acc2 = wmma_bf16(a, b2, acc2);
    acc3 = wmma_bf16(a, b3, acc3);
  }

  const float bb0 = bias[n0 + l16];
  const float bb1 = bias[n0 + 16 + l16];
  const float bb2 = bias[n0 + 32 + l16];
  const float bb3 = bias[n0 + 48 + l16];
#pragma unroll
  for (int j = 0; j < 8; ++j) {
    float* op = out + (size_t)(m0 + half * 8 + j) * DMODEL + n0 + l16;
    op[0]  = acc0[j] + bb0;
    op[16] = acc1[j] + bb1;
    op[32] = acc2[j] + bb2;
    op[48] = acc3[j] + bb3;
  }
}

// ---------------------------------------------------------------------------
extern "C" void kernel_launch(void* const* d_in, const int* in_sizes, int n_in,
                              void* d_out, int out_size, void* d_ws, size_t ws_size,
                              hipStream_t stream) {
  const float* q  = (const float*)d_in[0];
  const float* k  = (const float*)d_in[1];
  const float* v  = (const float*)d_in[2];
  const float* Wo = (const float*)d_in[3];
  const float* bo = (const float*)d_in[4];
  float* out = (float*)d_out;

  __bf16* Wb = (__bf16*)d_ws;                                        // 2 MB
  __bf16* X  = (__bf16*)((char*)d_ws + (size_t)DMODEL * DMODEL * 2); // 16 MB

  wcvt_kernel<<<(DMODEL * DMODEL) / (256 * 4), 256, 0, stream>>>(Wo, Wb);
  flash_kernel<<<dim3(SEQ / 128, BATCH * HEADS), 256, 0, stream>>>(q, k, v, X);
  proj_kernel<<<dim3((BATCH * SEQ) / 128, DMODEL / 64), 256, 0, stream>>>(X, Wb, bo, out);
}
